// ECGARD_44702019617306
// MI455X (gfx1250) — matrix operand order repaired and checked by reading, not compiled, not run on hardware
//
#include <hip/hip_runtime.h>
#include <hip/hip_bf16.h>
#include <math.h>

// ---------------- problem constants (match reference) ----------------
#define B_   4
#define L_   4096
#define D_   128          // d_model
#define E_   256          // d_inner
#define N_   16           // d_state
#define R_   8            // dt_rank
#define TOK_ (B_ * L_)    // 16384 tokens
#define KC_  (3 * D_)     // block conv flattened K = 384
#define LP_  (L_ + 2)     // padded sequence length (2 zero rows in front)
#define NCH_ 8            // scan chunks
#define LCH_ (L_ / NCH_)  // 512 steps per chunk

typedef __attribute__((ext_vector_type(16))) _Float16 v16h;
typedef __attribute__((ext_vector_type(8)))  _Float16 v8h;
typedef __attribute__((ext_vector_type(4)))  _Float16 v4h;
typedef __attribute__((ext_vector_type(8)))  float    v8f;
typedef __attribute__((ext_vector_type(4)))  float    v4f;

static __device__ __forceinline__ int lane_id() { return (int)(threadIdx.x & 31u); }

static __device__ __forceinline__ float silu_f(float v) { return v / (1.0f + __expf(-v)); }
static __device__ __forceinline__ float gelu_f(float v) { return 0.5f * v * (1.0f + erff(v * 0.70710678118654752f)); }

// ---------------- WMMA fragment loaders ----------------
// A fragment, 16-bit A 16x32 ISA layout. `row` = this lane's A row (M = lane&15).
static __device__ __forceinline__ v16h load_a_row(const _Float16* row, int kk) {
  const int kb = (lane_id() < 16) ? 0 : 8;
  const v8h lo = *(const v8h*)(row + kk + kb);
  const v8h hi = *(const v8h*)(row + kk + kb + 16);
  v16h a;
#pragma unroll
  for (int j = 0; j < 8; ++j) { a[j] = lo[j]; a[j + 8] = hi[j]; }
  return a;
}

// B fragment from an LDS-staged B^T band (row-padded to `ld` halves).
static __device__ __forceinline__ v16h load_bT_lds(const _Float16* sB, int n, int kk, int ld) {
  const int kb = (lane_id() < 16) ? 0 : 8;
  const _Float16* p = sB + (size_t)n * ld + kk + kb;
  const v8h lo = *(const v8h*)p;
  const v8h hi = *(const v8h*)(p + 16);
  v16h b;
#pragma unroll
  for (int j = 0; j < 8; ++j) { b[j] = lo[j]; b[j + 8] = hi[j]; }
  return b;
}

// Conv A fragment from padded activations hp (B, LP_, D_). rowbase = b*LP_ + tloc.
// 8-element K runs never cross a 128 boundary -> ktap constant per run, loads unconditional.
static __device__ __forceinline__ v16h load_a_conv(const _Float16* hp, int rowbase, int kk) {
  const int kb  = (lane_id() < 16) ? 0 : 8;
  const int kg0 = kk + kb;
  const int kt0 = kg0 >> 7, i0 = kg0 & 127;
  const int kg1 = kg0 + 16;
  const int kt1 = kg1 >> 7, i1 = kg1 & 127;
  const v8h lo = *(const v8h*)(hp + (size_t)(rowbase + kt0) * D_ + i0);
  const v8h hi = *(const v8h*)(hp + (size_t)(rowbase + kt1) * D_ + i1);
  v16h a;
#pragma unroll
  for (int j = 0; j < 8; ++j) { a[j] = lo[j]; a[j + 8] = hi[j]; }
  return a;
}

// =====================================================================
// 0) One-shot pack: weights -> f16 transposed (N x K) + zero hp pad rows
// =====================================================================
#define PK_CW  (D_ * KC_)          // 49152
#define PK_IN  (2 * E_ * D_)       // 65536
#define PK_OUT (D_ * E_)           // 32768
#define PK_PAD (B_ * 2 * D_)       // 1024
#define PK_TOT (PK_CW + PK_IN + PK_OUT + PK_PAD)   // 148480

__global__ void k_pack(const float* __restrict__ conv_w, const float* __restrict__ in_w,
                       const float* __restrict__ out_w, _Float16* __restrict__ cwT,
                       _Float16* __restrict__ inwT, _Float16* __restrict__ outwT,
                       _Float16* __restrict__ hp) {
  const int idx = (int)(blockIdx.x * blockDim.x + threadIdx.x);
  if (idx < PK_CW) {                         // cwT[o][kg], kg = ktap*128 + i
    const int o = idx / KC_, kg = idx % KC_;
    const int kt = kg >> 7, i = kg & 127;
    cwT[idx] = (_Float16)conv_w[(size_t)o * KC_ + i * 3 + kt];
  } else if (idx < PK_CW + PK_IN) {          // inwT[n][k] = in_w[k][n]
    const int j = idx - PK_CW;
    const int n = j >> 7, k = j & 127;
    inwT[j] = (_Float16)in_w[(size_t)k * (2 * E_) + n];
  } else if (idx < PK_CW + PK_IN + PK_OUT) { // outwT[n][k] = out_w[k][n]
    const int j = idx - (PK_CW + PK_IN);
    const int n = j >> 8, k = j & 255;
    outwT[j] = (_Float16)out_w[(size_t)k * D_ + n];
  } else {                                   // zero the 2 pad rows per batch
    const int j = idx - (PK_CW + PK_IN + PK_OUT);
    const int bq = j >> 8, r = j & 255;
    hp[(size_t)bq * LP_ * D_ + r] = (_Float16)0.0f;
  }
}

// =====================================================================
// 1) LayerNorm -> f16 padded activations. One wave/token, float4 loads.
// =====================================================================
__global__ void k_layernorm(const float* __restrict__ x, const float* __restrict__ g,
                            const float* __restrict__ b, _Float16* __restrict__ hp) {
  const int tok  = (int)((blockIdx.x * blockDim.x + threadIdx.x) >> 5);
  const int lane = lane_id();
  const int bidx = tok >> 12;
  const int tloc = tok & (L_ - 1);
  const v4f xv = ((const v4f*)(x + (size_t)tok * D_))[lane];
  float s = 0.f, s2 = 0.f;
#pragma unroll
  for (int j = 0; j < 4; ++j) { s += xv[j]; s2 += xv[j] * xv[j]; }
#pragma unroll
  for (int m = 16; m >= 1; m >>= 1) { s += __shfl_xor(s, m); s2 += __shfl_xor(s2, m); }
  const float mu  = s * (1.0f / D_);
  const float var = s2 * (1.0f / D_) - mu * mu;
  const float rr  = rsqrtf(var + 1e-5f);
  const v4f gv = ((const v4f*)g)[lane];
  const v4f bv = ((const v4f*)b)[lane];
  v4h o;
#pragma unroll
  for (int j = 0; j < 4; ++j) o[j] = (_Float16)((xv[j] - mu) * rr * gv[j] + bv[j]);
  ((v4h*)(hp + ((size_t)bidx * LP_ + tloc + 2) * D_))[lane] = o;
}

// =====================================================================
// 2) Block causal conv as WMMA. Wave = 16x128 row-band (8 acc tiles).
//    Weights LDS-staged in 2 K-phases of 192 (51 KB, row pad +8 halves).
// =====================================================================
#define CONV_LD  (192 + 8)   // halves per staged row
__global__ void k_conv_gemm(const _Float16* __restrict__ hp, const _Float16* __restrict__ cwT,
                            const float* __restrict__ cb, float* __restrict__ h_conv) {
  __shared__ alignas(16) _Float16 sB[D_ * CONV_LD];   // 51,200 B
  const int tm   = (int)((blockIdx.x * blockDim.x + threadIdx.x) >> 5);  // 1024 token tiles
  const int lane = lane_id();
  const int mrow = tm * 16 + (lane & 15);
  const int tloc = mrow & (L_ - 1);
  const int bidx = mrow >> 12;
  const int rowbase = bidx * LP_ + tloc;
  const int nl = lane & 15;

  v8f acc[8] = {};
#pragma unroll
  for (int half = 0; half < 2; ++half) {
    // cooperative stage of K band [half*192, half*192+192) for all 128 rows
    for (int idx = (int)threadIdx.x; idx < D_ * 24; idx += 256) {
      const int r = idx / 24, c8 = idx % 24;
      *(v8h*)(sB + (size_t)r * CONV_LD + c8 * 8) =
          *(const v8h*)(cwT + (size_t)r * KC_ + half * 192 + c8 * 8);
    }
    __syncthreads();
#pragma unroll
    for (int kk = 0; kk < 192; kk += 32) {
      const v16h af = load_a_conv(hp, rowbase, half * 192 + kk);
#pragma unroll
      for (int tn = 0; tn < 8; ++tn) {
        const v16h bf = load_bT_lds(sB, tn * 16 + nl, kk, CONV_LD);
        acc[tn] = __builtin_amdgcn_wmma_f32_16x16x32_f16(false, af, false, bf, (short)0, acc[tn], false, false);
      }
    }
    __syncthreads();
  }
  const int mo = (lane < 16) ? 0 : 8;
#pragma unroll
  for (int tn = 0; tn < 8; ++tn) {
    const int nCol = tn * 16 + nl;
    const float bias = cb[nCol];
#pragma unroll
    for (int v = 0; v < 8; ++v)
      h_conv[(size_t)(tm * 16 + v + mo) * D_ + nCol] = gelu_f(acc[tn][v] + bias);
  }
}

// =====================================================================
// 3) in_proj: (16384x128)@(128x512)+bias -> xz. Wave = 16x128 row-band.
//    Waves of a block share one 128-wide N band -> one LDS stage (35 KB).
// =====================================================================
#define IN_LD (D_ + 8)
__global__ void k_inproj(const _Float16* __restrict__ hp, const _Float16* __restrict__ inwT,
                         const float* __restrict__ bias, float* __restrict__ xz) {
  __shared__ alignas(16) _Float16 sB[D_ * IN_LD];     // 34,816 B
  const int wid  = (int)((blockIdx.x * blockDim.x + threadIdx.x) >> 5);  // 4096 waves
  const int lane = lane_id();
  const int nb = wid >> 10;                  // 128-wide N band (uniform per block)
  const int tm = wid & 1023;
  const int mrow = tm * 16 + (lane & 15);
  const int tloc = mrow & (L_ - 1);
  const int bidx = mrow >> 12;
  const _Float16* arow = hp + ((size_t)bidx * LP_ + tloc + 2) * D_;
  const int nl = lane & 15;

  // stage this block's weight band (rows nb*128 .. nb*128+127, K=128)
  for (int idx = (int)threadIdx.x; idx < D_ * 16; idx += 256) {
    const int r = idx / 16, c8 = idx % 16;
    *(v8h*)(sB + (size_t)r * IN_LD + c8 * 8) =
        *(const v8h*)(inwT + (size_t)(nb * D_ + r) * D_ + c8 * 8);
  }
  __syncthreads();

  v8f acc[8] = {};
#pragma unroll
  for (int kk = 0; kk < D_; kk += 32) {
    const v16h af = load_a_row(arow, kk);
#pragma unroll
    for (int tn = 0; tn < 8; ++tn) {
      const v16h bf = load_bT_lds(sB, tn * 16 + nl, kk, IN_LD);
      acc[tn] = __builtin_amdgcn_wmma_f32_16x16x32_f16(false, af, false, bf, (short)0, acc[tn], false, false);
    }
  }
  const int mo = (lane < 16) ? 0 : 8;
#pragma unroll
  for (int tn = 0; tn < 8; ++tn) {
    const int n = (nb * 8 + tn) * 16 + nl;
    const float bv = bias[n];
#pragma unroll
    for (int v = 0; v < 8; ++v)
      xz[(size_t)(tm * 16 + v + mo) * (2 * E_) + n] = acc[tn][v] + bv;
  }
}

// =====================================================================
// 4) Depthwise causal conv1d (K=4) + SiLU on x_in half of xz -> x_c
// =====================================================================
__global__ void k_dwconv(const float* __restrict__ xz, const float* __restrict__ w,
                         const float* __restrict__ b, float* __restrict__ x_c) {
  const int idx = (int)(blockIdx.x * blockDim.x + threadIdx.x);
  const int e   = idx & (E_ - 1);
  const int tok = idx >> 8;
  const int tloc = tok & (L_ - 1);
  float acc = b[e];
#pragma unroll
  for (int k = 0; k < 4; ++k) {
    const int tt = tloc - 3 + k;
    if (tt >= 0) acc += w[e * 4 + k] * xz[(size_t)(tok - 3 + k) * (2 * E_) + e];
  }
  x_c[(size_t)tok * E_ + e] = silu_f(acc);
}

// =====================================================================
// 5) x_proj (256->40) + dt_proj (8->256) + softplus. Wave per token.
// =====================================================================
__global__ void k_xproj_dt(const float* __restrict__ x_c, const float* __restrict__ Wx,
                           const float* __restrict__ Wdt, const float* __restrict__ bdt,
                           float* __restrict__ dt, float* __restrict__ Bm, float* __restrict__ Cm) {
  __shared__ float sx[8][E_];
  __shared__ float sd[8][48];
  const int wv   = (int)(threadIdx.x >> 5);
  const int lane = lane_id();
  const int tok  = (int)blockIdx.x * 8 + wv;
  const float* xr = x_c + (size_t)tok * E_;
#pragma unroll
  for (int j = 0; j < 8; ++j) sx[wv][lane + 32 * j] = xr[lane + 32 * j];
  __syncthreads();
  for (int jj = lane; jj < R_ + 2 * N_; jj += 32) {
    float acc = 0.f;
#pragma unroll 8
    for (int i = 0; i < E_; ++i) acc += sx[wv][i] * Wx[(size_t)i * (R_ + 2 * N_) + jj];
    sd[wv][jj] = acc;
  }
  __syncthreads();
  if (lane < N_) {
    Bm[(size_t)tok * N_ + lane] = sd[wv][R_ + lane];
    Cm[(size_t)tok * N_ + lane] = sd[wv][R_ + N_ + lane];
  }
#pragma unroll
  for (int j = 0; j < 8; ++j) {
    const int e = lane + 32 * j;
    float acc = bdt[e];
#pragma unroll
    for (int r = 0; r < R_; ++r) acc += sd[wv][r] * Wdt[(size_t)r * E_ + e];
    dt[(size_t)tok * E_ + e] = (acc > 20.f) ? acc : log1pf(__expf(acc));   // softplus
  }
}

// =====================================================================
// 6) Chunk-parallel selective scan (linear recurrence h = dA*h + dBu):
//    pass1: per-chunk hEnd (h0=0) and P = prod(dA)    [131072 thr]
//    pass2: h0[c] = P[c-1]*h0[c-1] + hEnd[c-1]        [16384 thr, 8 steps]
//    pass3: recompute chunk from true h0, emit gated y [131072 thr]
//    Thread -> (b,ch,e,n); n in low 4 bits so the sum over n is 4 shfl_xor.
// =====================================================================
__global__ void k_scan1(const float* __restrict__ dt, const float* __restrict__ Bm,
                        const float* __restrict__ x_c, const float* __restrict__ A_log,
                        float* __restrict__ hEnd, float* __restrict__ Pprod) {
  const int tid = (int)(blockIdx.x * blockDim.x + threadIdx.x);
  const int n  = tid & (N_ - 1);
  const int e  = (tid >> 4) & (E_ - 1);
  const int ch = (tid >> 12) & (NCH_ - 1);
  const int b  = tid >> 15;
  const float A = -__expf(A_log[e * N_ + n]);
  float h = 0.f, P = 1.f;
  const size_t tok0 = (size_t)b * L_ + (size_t)ch * LCH_;
  for (int t = 0; t < LCH_; t += 4) {
    float dtv[4], u[4], Bv[4];
#pragma unroll
    for (int q = 0; q < 4; ++q) {
      const size_t tok = tok0 + t + q;
      dtv[q] = dt[tok * E_ + e];
      u[q]   = x_c[tok * E_ + e];
      Bv[q]  = Bm[tok * N_ + n];
    }
#pragma unroll
    for (int q = 0; q < 4; ++q) {
      const float dA = __expf(dtv[q] * A);
      h = dA * h + (dtv[q] * Bv[q]) * u[q];
      P *= dA;
    }
  }
  hEnd[tid] = h;
  Pprod[tid] = P;
}

__global__ void k_scan2(const float* __restrict__ hEnd, const float* __restrict__ Pprod,
                        float* __restrict__ h0) {
  const int tid = (int)(blockIdx.x * blockDim.x + threadIdx.x);   // (b,e,n)
  const int n = tid & (N_ - 1);
  const int e = (tid >> 4) & (E_ - 1);
  const int b = tid >> 12;
  float h = 0.f;
#pragma unroll
  for (int ch = 0; ch < NCH_; ++ch) {
    const size_t idx = ((((size_t)b * NCH_ + ch) * E_) + e) * N_ + n;
    h0[idx] = h;
    h = Pprod[idx] * h + hEnd[idx];
  }
}

__global__ void k_scan3(const float* __restrict__ dt, const float* __restrict__ Bm,
                        const float* __restrict__ Cm, const float* __restrict__ x_c,
                        const float* __restrict__ xz, const float* __restrict__ A_log,
                        const float* __restrict__ Dskip, const float* __restrict__ h0,
                        _Float16* __restrict__ y_g) {
  const int tid = (int)(blockIdx.x * blockDim.x + threadIdx.x);
  const int n  = tid & (N_ - 1);
  const int e  = (tid >> 4) & (E_ - 1);
  const int ch = (tid >> 12) & (NCH_ - 1);
  const int b  = tid >> 15;
  const float A  = -__expf(A_log[e * N_ + n]);
  const float Dv = Dskip[e];
  float h = h0[tid];
  const size_t tok0 = (size_t)b * L_ + (size_t)ch * LCH_;
  for (int t = 0; t < LCH_; t += 4) {
    float dtv[4], u[4], Bv[4], Cv[4], z[4];
#pragma unroll
    for (int q = 0; q < 4; ++q) {
      const size_t tok = tok0 + t + q;
      dtv[q] = dt[tok * E_ + e];
      u[q]   = x_c[tok * E_ + e];
      Bv[q]  = Bm[tok * N_ + n];
      Cv[q]  = Cm[tok * N_ + n];
      z[q]   = xz[tok * (2 * E_) + E_ + e];
    }
#pragma unroll
    for (int q = 0; q < 4; ++q) {
      const float dA = __expf(dtv[q] * A);
      h = dA * h + (dtv[q] * Bv[q]) * u[q];
      float p = h * Cv[q];
      p += __shfl_xor(p, 1); p += __shfl_xor(p, 2);
      p += __shfl_xor(p, 4); p += __shfl_xor(p, 8);
      if (n == 0) {
        const float y = p + u[q] * Dv;
        y_g[(tok0 + t + q) * E_ + e] = (_Float16)(y * silu_f(z[q]));
      }
    }
  }
}

// =====================================================================
// 7) out_proj + fused residual: out = x + 0.5*(y_g@W + b) + 0.5*h_conv
//    Weights LDS-staged in 2 K-phases of 128 (35 KB).
// =====================================================================
#define OUT_LD (D_ + 8)
__global__ void k_outproj(const _Float16* __restrict__ yg, const _Float16* __restrict__ outwT,
                          const float* __restrict__ bias, const float* __restrict__ x,
                          const float* __restrict__ h_conv, float* __restrict__ out) {
  __shared__ alignas(16) _Float16 sB[D_ * OUT_LD];    // 34,816 B
  const int tm   = (int)((blockIdx.x * blockDim.x + threadIdx.x) >> 5);  // 1024
  const int lane = lane_id();
  const int m  = tm * 16 + (lane & 15);
  const _Float16* arow = yg + (size_t)m * E_;
  const int nl = lane & 15;

  v8f acc[8] = {};
#pragma unroll
  for (int half = 0; half < 2; ++half) {
    for (int idx = (int)threadIdx.x; idx < D_ * 16; idx += 256) {
      const int r = idx / 16, c8 = idx % 16;
      *(v8h*)(sB + (size_t)r * OUT_LD + c8 * 8) =
          *(const v8h*)(outwT + (size_t)r * E_ + half * D_ + c8 * 8);
    }
    __syncthreads();
#pragma unroll
    for (int kk = 0; kk < D_; kk += 32) {
      const v16h af = load_a_row(arow, half * D_ + kk);
#pragma unroll
      for (int tn = 0; tn < 8; ++tn) {
        const v16h bf = load_bT_lds(sB, tn * 16 + nl, kk, OUT_LD);
        acc[tn] = __builtin_amdgcn_wmma_f32_16x16x32_f16(false, af, false, bf, (short)0, acc[tn], false, false);
      }
    }
    __syncthreads();
  }
  const int mo = (lane < 16) ? 0 : 8;
#pragma unroll
  for (int tn = 0; tn < 8; ++tn) {
    const int n = tn * 16 + nl;
    const float bv = bias[n];
#pragma unroll
    for (int v = 0; v < 8; ++v) {
      const size_t o = (size_t)(tm * 16 + v + mo) * D_ + n;
      out[o] = x[o] + 0.5f * (acc[tn][v] + bv) + 0.5f * h_conv[o];
    }
  }
}

// =====================================================================
extern "C" void kernel_launch(void* const* d_in, const int* in_sizes, int n_in,
                              void* d_out, int out_size, void* d_ws, size_t ws_size,
                              hipStream_t stream) {
  const float* x      = (const float*)d_in[0];
  const float* ln_g   = (const float*)d_in[1];
  const float* ln_b   = (const float*)d_in[2];
  const float* conv_w = (const float*)d_in[3];
  const float* conv_b = (const float*)d_in[4];
  const float* in_w   = (const float*)d_in[5];
  const float* in_b   = (const float*)d_in[6];
  const float* c1_w   = (const float*)d_in[7];
  const float* c1_b   = (const float*)d_in[8];
  const float* xp_w   = (const float*)d_in[9];
  const float* dtp_w  = (const float*)d_in[10];
  const float* dtp_b  = (const float*)d_in[11];
  const float* A_log  = (const float*)d_in[12];
  const float* Dskip  = (const float*)d_in[13];
  const float* out_w  = (const float*)d_in[14];
  const float* out_b  = (const float*)d_in[15];
  float* out = (float*)d_out;

  // workspace carve-up (~92 MB; fully L2-resident on MI455X's 192 MB L2)
  char*  base = (char*)d_ws;
  size_t off  = 0;
  auto carve = [&](size_t bytes) -> void* {
    void* p = base + off;
    off = (off + bytes + 255) & ~(size_t)255;
    return p;
  };
  _Float16* hp     = (_Float16*)carve((size_t)B_ * LP_ * D_ * 2);
  _Float16* cwT    = (_Float16*)carve((size_t)PK_CW * 2);
  _Float16* inwT   = (_Float16*)carve((size_t)PK_IN * 2);
  _Float16* outwT  = (_Float16*)carve((size_t)PK_OUT * 2);
  _Float16* y_g    = (_Float16*)carve((size_t)TOK_ * E_ * 2);
  float*    h_conv = (float*)carve((size_t)TOK_ * D_ * 4);
  float*    xz     = (float*)carve((size_t)TOK_ * 2 * E_ * 4);
  float*    x_c    = (float*)carve((size_t)TOK_ * E_ * 4);
  float*    dt     = (float*)carve((size_t)TOK_ * E_ * 4);
  float*    Bmat   = (float*)carve((size_t)TOK_ * N_ * 4);
  float*    Cmat   = (float*)carve((size_t)TOK_ * N_ * 4);
  float*    hEnd   = (float*)carve((size_t)B_ * NCH_ * E_ * N_ * 4);
  float*    Pprod  = (float*)carve((size_t)B_ * NCH_ * E_ * N_ * 4);
  float*    h0     = (float*)carve((size_t)B_ * NCH_ * E_ * N_ * 4);

  k_pack     <<<PK_TOT / 256, 256, 0, stream>>>(conv_w, in_w, out_w, cwT, inwT, outwT, hp);
  k_layernorm<<<TOK_ / 8, 256, 0, stream>>>(x, ln_g, ln_b, hp);
  k_conv_gemm<<<(TOK_ / 16) / 8, 256, 0, stream>>>(hp, cwT, conv_b, h_conv);
  k_inproj   <<<(TOK_ / 16) * 4 / 8, 256, 0, stream>>>(hp, inwT, in_b, xz);
  k_dwconv   <<<(TOK_ * E_) / 256, 256, 0, stream>>>(xz, c1_w, c1_b, x_c);
  k_xproj_dt <<<TOK_ / 8, 256, 0, stream>>>(x_c, xp_w, dtp_w, dtp_b, dt, Bmat, Cmat);
  k_scan1    <<<(B_ * NCH_ * E_ * N_) / 256, 256, 0, stream>>>(dt, Bmat, x_c, A_log, hEnd, Pprod);
  k_scan2    <<<(B_ * E_ * N_) / 256, 256, 0, stream>>>(hEnd, Pprod, h0);
  k_scan3    <<<(B_ * NCH_ * E_ * N_) / 256, 256, 0, stream>>>(dt, Bmat, Cmat, x_c, xz, A_log, Dskip, h0, y_g);
  k_outproj  <<<(TOK_ / 16) / 8, 256, 0, stream>>>(y_g, outwT, out_b, x, h_conv, out);
}
